// ModifiedMultiHeadedAttention_6760278524293
// MI455X (gfx1250) — matrix operand order repaired and checked by reading, not compiled, not run on hardware
//
#include <hip/hip_runtime.h>
#include <hip/hip_bf16.h>

// MI455X fused attention: f32 -> bf16, WMMA 16x16x32 bf16 everywhere,
// flash-style fused softmax (never materialize [B,H,T,T]).
// Double-buffered LDS staging; async global->LDS copies when the toolchain
// exposes the gfx1250 builtins (guarded, with a register-copy fallback).

typedef __attribute__((ext_vector_type(16))) __bf16 v16bf;
typedef __attribute__((ext_vector_type(8)))  __bf16 v8bf;
typedef __attribute__((ext_vector_type(8)))  float  v8f;
typedef int v4i __attribute__((vector_size(4 * sizeof(int))));

#define NUM_HEADS 16
#define SEQ_T     2048
#define DMODEL    1024
#define DHEAD     64

// ---- gfx1250 async global->LDS path (ASYNCcnt), guarded ------------------
#if defined(__has_builtin)
#if __has_builtin(__builtin_amdgcn_global_load_async_to_lds_b128) && \
    __has_builtin(__builtin_amdgcn_s_wait_asynccnt)
#define HAVE_ASYNC_LDS 1
#endif
#endif

#if defined(HAVE_ASYNC_LDS)
__device__ __forceinline__ void async_copy16(__bf16* lds, const __bf16* g) {
  // signature (from compiler diagnostic): param0 = int4 addrspace(1)*,
  // then LDS pointer, imm offset, imm cpol.
  __builtin_amdgcn_global_load_async_to_lds_b128(
      (__attribute__((address_space(1))) v4i*)g,
      (__attribute__((address_space(3))) v4i*)lds, 0, 0);
}
__device__ __forceinline__ void wait_async() {
  __builtin_amdgcn_s_wait_asynccnt(0);
}
#else
__device__ __forceinline__ void async_copy16(__bf16* lds, const __bf16* g) {
  *(v8bf*)lds = *(const v8bf*)g;
}
__device__ __forceinline__ void wait_async() {}
#endif

__device__ __forceinline__ v8f wmma_bf16(v16bf a, v16bf b, v8f c) {
  return __builtin_amdgcn_wmma_f32_16x16x32_bf16(false, a, false, b, (short)0, c,
                                                 false, false);
}

// A-fragment (16x32 bf16, MxK): lane L holds row m=L&15; half kh=L>>4.
// Register pairs 0..3 -> K = kh*8 + {0..7}; pairs 4..7 -> K = 16 + kh*8 + {0..7}.
__device__ __forceinline__ v16bf load_frag_a(const __bf16* base, int ld) {
  int lane = threadIdx.x & 31;
  int m  = lane & 15;
  int kh = (lane >> 4) & 1;
  const __bf16* p = base + m * ld + kh * 8;
  v16bf f;
#pragma unroll
  for (int i = 0; i < 8; ++i) { f[i] = p[i]; f[8 + i] = p[16 + i]; }
  return f;
}

// B-fragment (32x16 bf16, KxN) from an LDS tile stored TRANSPOSED as [N][K]:
// lane L holds col n=L&15; half kh=L>>4; K = kh*8 + {0..7}, 16 + kh*8 + {0..7}.
__device__ __forceinline__ v16bf load_frag_b(const __bf16* baseT, int ld) {
  int lane = threadIdx.x & 31;
  int n  = lane & 15;
  int kh = (lane >> 4) & 1;
  const __bf16* p = baseT + n * ld + kh * 8;
  v16bf f;
#pragma unroll
  for (int i = 0; i < 8; ++i) { f[i] = p[i]; f[8 + i] = p[16 + i]; }
  return f;
}

// ---------------------------------------------------------------- converts
__global__ void f32_to_bf16_kernel(const float* __restrict__ src,
                                   __bf16* __restrict__ dst, int n) {
  int i = blockIdx.x * blockDim.x + threadIdx.x;
  if (i < n) dst[i] = (__bf16)src[i];
}

// ---------------------------------------------------------------- GEMM core
// Block tile 128(M) x 128(N), K-step 32, double-buffered LDS (one barrier
// per K-step; next tile's global traffic overlaps current tile's WMMAs).
// 8 waves: 4(M) x 2(N), each wave owns 32x64 = 2x4 WMMA accumulators.
#define TILE_ELEMS (128 * 32)

__device__ __forceinline__ void gemm_mainloop(const __bf16* __restrict__ A,
                                              const __bf16* __restrict__ Bm,
                                              int K, int N, int bm0, int bn0,
                                              __bf16* As,   // [2][128*32]
                                              __bf16* BsT,  // [2][128*32]
                                              v8f acc[2][4]) {
  int t = threadIdx.x;
  int wave = t >> 5;
  int wm = wave >> 1;                  // 0..3
  int wn = wave & 1;                   // 0..1
  int arow = t >> 1, ac0 = (t & 1) * 16;
  int bk = t >> 3, bcol = (t & 7) * 16;
  const __bf16* gA = A + (size_t)(bm0 + arow) * K + ac0;
  const __bf16* gB = Bm + (size_t)bk * N + (bn0 + bcol);

  // prologue: stage k0 = 0 into buffer 0
  async_copy16(As + arow * 32 + ac0, gA);
  {
    v8bf r0 = ((const v8bf*)gB)[0], r1 = ((const v8bf*)gB)[1];
#pragma unroll
    for (int i = 0; i < 8; ++i) {
      BsT[(bcol + i) * 32 + bk]     = r0[i];
      BsT[(bcol + 8 + i) * 32 + bk] = r1[i];
    }
  }

  int cur = 0;
  for (int k0 = 0; k0 < K; k0 += 32) {
    wait_async();
    __syncthreads();  // buf[cur] fully staged; buf[cur^1] free to overwrite
    bool more = (k0 + 32) < K;
    v8bf r0, r1;
    if (more) {  // issue next tile's global traffic before computing
      async_copy16(As + (cur ^ 1) * TILE_ELEMS + arow * 32 + ac0,
                   gA + (k0 + 32));
      const v8bf* g = (const v8bf*)(gB + (size_t)(k0 + 32) * N);
      r0 = g[0];
      r1 = g[1];
    }
    const __bf16* Ac = As + cur * TILE_ELEMS;
    const __bf16* Bc = BsT + cur * TILE_ELEMS;
    v16bf a0 = load_frag_a(Ac + (wm * 32) * 32, 32);
    v16bf a1 = load_frag_a(Ac + (wm * 32 + 16) * 32, 32);
#pragma unroll
    for (int tn = 0; tn < 4; ++tn) {
      v16bf bf = load_frag_b(Bc + (wn * 64 + tn * 16) * 32, 32);
      acc[0][tn] = wmma_bf16(a0, bf, acc[0][tn]);
      acc[1][tn] = wmma_bf16(a1, bf, acc[1][tn]);
    }
    if (more) {  // transpose-store next B tile into the other buffer
      __bf16* Bn = BsT + (cur ^ 1) * TILE_ELEMS;
#pragma unroll
      for (int i = 0; i < 8; ++i) {
        Bn[(bcol + i) * 32 + bk]     = r0[i];
        Bn[(bcol + 8 + i) * 32 + bk] = r1[i];
      }
    }
    cur ^= 1;
  }
}

// qkv projection: [4096,1024] x [1024,3072] + bias, scatter into
// q/k/v buffers laid out [B*H, T, 64] bf16.
__global__ __launch_bounds__(256) void qkv_gemm_kernel(
    const __bf16* __restrict__ A, const __bf16* __restrict__ Bm,
    const float* __restrict__ bias, __bf16* __restrict__ qb,
    __bf16* __restrict__ kb, __bf16* __restrict__ vb) {
  __shared__ __attribute__((aligned(16))) __bf16 As[2 * TILE_ELEMS];
  __shared__ __attribute__((aligned(16))) __bf16 BsT[2 * TILE_ELEMS];
  const int K = DMODEL, N = 3 * DMODEL;
  int bm0 = blockIdx.y * 128, bn0 = blockIdx.x * 128;
  v8f acc[2][4] = {};
  gemm_mainloop(A, Bm, K, N, bm0, bn0, As, BsT, acc);

  int t = threadIdx.x, wave = t >> 5, lane = t & 31;
  int wm = wave >> 1, wn = wave & 1;
  int hl = lane >> 4, nlo = lane & 15;
#pragma unroll
  for (int tm = 0; tm < 2; ++tm)
#pragma unroll
    for (int tn = 0; tn < 4; ++tn)
#pragma unroll
      for (int r = 0; r < 8; ++r) {
        int m = bm0 + wm * 32 + tm * 16 + r + 8 * hl;   // row in [B*T]
        int e = bn0 + wn * 64 + tn * 16 + nlo;          // col in [3D]
        float v = acc[tm][tn][r] + bias[e];
        int part = e >> 10;        // 0=q 1=k 2=v
        int d = e & 1023;
        int hh = d >> 6, j = d & 63;
        int bb = m >> 11, tt = m & 2047;
        __bf16* dst = (part == 0) ? qb : ((part == 1) ? kb : vb);
        dst[((size_t)(bb * NUM_HEADS + hh) * SEQ_T + tt) * DHEAD + j] = (__bf16)v;
      }
}

// output projection: [4096,1024] x [1024,1024] + bias -> f32 out
__global__ __launch_bounds__(256) void proj_gemm_kernel(
    const __bf16* __restrict__ A, const __bf16* __restrict__ Bm,
    const float* __restrict__ bias, float* __restrict__ out) {
  __shared__ __attribute__((aligned(16))) __bf16 As[2 * TILE_ELEMS];
  __shared__ __attribute__((aligned(16))) __bf16 BsT[2 * TILE_ELEMS];
  const int K = DMODEL, N = DMODEL;
  int bm0 = blockIdx.y * 128, bn0 = blockIdx.x * 128;
  v8f acc[2][4] = {};
  gemm_mainloop(A, Bm, K, N, bm0, bn0, As, BsT, acc);

  int t = threadIdx.x, wave = t >> 5, lane = t & 31;
  int wm = wave >> 1, wn = wave & 1;
  int hl = lane >> 4, nlo = lane & 15;
#pragma unroll
  for (int tm = 0; tm < 2; ++tm)
#pragma unroll
    for (int tn = 0; tn < 4; ++tn)
#pragma unroll
      for (int r = 0; r < 8; ++r) {
        int m = bm0 + wm * 32 + tm * 16 + r + 8 * hl;
        int e = bn0 + wn * 64 + tn * 16 + nlo;
        out[(size_t)m * N + e] = acc[tm][tn][r] + bias[e];
      }
}

// ---------------------------------------------------------------- flash attn
// Block: 128 query rows of one (b,h); 8 waves x 16 q-rows each.
// Key/value tiles of 64 staged in LDS; K tile via async global->LDS, V tile
// transposed through registers so the P@V B-fragment load is contiguous.
__global__ __launch_bounds__(256) void flash_attn_kernel(
    const __bf16* __restrict__ qb, const __bf16* __restrict__ kb,
    const __bf16* __restrict__ vb, __bf16* __restrict__ attn) {
  __shared__ __attribute__((aligned(16))) __bf16 Ks[64 * 64];   // [key][dh]
  __shared__ __attribute__((aligned(16))) __bf16 VsT[64 * 64];  // [dh][key]
  __shared__ __attribute__((aligned(16))) __bf16 Ps[8 * 16 * 64];

  int t = threadIdx.x, wave = t >> 5, lane = t & 31;
  int hl = lane >> 4, nlo = lane & 15;
  int bh = blockIdx.y;                     // b*H + h
  int qw0 = blockIdx.x * 128 + wave * 16;  // wave's first query row
  const __bf16* Q  = qb + (size_t)bh * SEQ_T * DHEAD;
  const __bf16* Kp = kb + (size_t)bh * SEQ_T * DHEAD;
  const __bf16* Vp = vb + (size_t)bh * SEQ_T * DHEAD;

  // Q fragments (dh=64 -> two K-chunks of 32), straight from global.
  v16bf aQ0 = load_frag_a(Q + (size_t)qw0 * DHEAD, DHEAD);
  v16bf aQ1 = load_frag_a(Q + (size_t)qw0 * DHEAD + 32, DHEAD);

  v8f O[4] = {};
  float mrow[8], lrow[8];
#pragma unroll
  for (int r = 0; r < 8; ++r) { mrow[r] = -3.0e38f; lrow[r] = 0.f; }

  int nkt = blockIdx.x * 2 + 2;  // causal: key tiles of 64 up to block end
  for (int kt = 0; kt < nkt; ++kt) {
    int j0 = kt * 64;
    __syncthreads();
    int row = t >> 2, c0 = (t & 3) * 16;
    // K tile, natural [key][dh]: contiguous -> async global->LDS
    async_copy16(Ks + row * 64 + c0, Kp + (size_t)(j0 + row) * DHEAD + c0);
    {  // V tile, transposed into [dh][key] through registers
      const v8bf* g = (const v8bf*)(Vp + (size_t)(j0 + row) * DHEAD + c0);
      v8bf r0 = g[0], r1 = g[1];
#pragma unroll
      for (int i = 0; i < 8; ++i) {
        VsT[(c0 + i) * 64 + row]     = r0[i];
        VsT[(c0 + 8 + i) * 64 + row] = r1[i];
      }
    }
    wait_async();
    __syncthreads();

    // S = Q @ K^T  (16 x 64 per wave)
    v8f S[4];
#pragma unroll
    for (int nt = 0; nt < 4; ++nt) {
      v8f z = {};
      z = wmma_bf16(aQ0, load_frag_b(Ks + (nt * 16) * 64, 64), z);
      z = wmma_bf16(aQ1, load_frag_b(Ks + (nt * 16) * 64 + 32, 64), z);
      S[nt] = z;
    }
    // scale + causal mask (C layout: row = r + 8*hl, col = nlo)
#pragma unroll
    for (int nt = 0; nt < 4; ++nt)
#pragma unroll
      for (int r = 0; r < 8; ++r) {
        int qi = qw0 + r + 8 * hl;
        int ki = j0 + nt * 16 + nlo;
        float s = S[nt][r] * 0.125f;  // 1/sqrt(64)
        S[nt][r] = (ki > qi) ? -1.0e9f : s;
      }
    // online softmax row statistics (reduce across the 16 column lanes)
    float alpha[8], rs[8];
#pragma unroll
    for (int r = 0; r < 8; ++r) {
      float v = fmaxf(fmaxf(S[0][r], S[1][r]), fmaxf(S[2][r], S[3][r]));
      v = fmaxf(v, __shfl_xor(v, 1, 32));
      v = fmaxf(v, __shfl_xor(v, 2, 32));
      v = fmaxf(v, __shfl_xor(v, 4, 32));
      v = fmaxf(v, __shfl_xor(v, 8, 32));
      float mn = fmaxf(mrow[r], v);
      alpha[r] = __expf(mrow[r] - mn);
      mrow[r] = mn;
      rs[r] = 0.f;
    }
#pragma unroll
    for (int nt = 0; nt < 4; ++nt)
#pragma unroll
      for (int r = 0; r < 8; ++r) {
        float p = __expf(S[nt][r] - mrow[r]);
        S[nt][r] = p;
        rs[r] += p;
      }
#pragma unroll
    for (int r = 0; r < 8; ++r) {
      float v = rs[r];
      v += __shfl_xor(v, 1, 32);
      v += __shfl_xor(v, 2, 32);
      v += __shfl_xor(v, 4, 32);
      v += __shfl_xor(v, 8, 32);
      lrow[r] = lrow[r] * alpha[r] + v;
    }
#pragma unroll
    for (int jt = 0; jt < 4; ++jt)
#pragma unroll
      for (int r = 0; r < 8; ++r) O[jt][r] *= alpha[r];

    // re-layout P: C layout -> A fragments via per-wave LDS scratch
    __bf16* Pw = Ps + wave * (16 * 64);
#pragma unroll
    for (int nt = 0; nt < 4; ++nt)
#pragma unroll
      for (int r = 0; r < 8; ++r)
        Pw[(r + 8 * hl) * 64 + nt * 16 + nlo] = (__bf16)S[nt][r];
    __syncthreads();
    v16bf aP0 = load_frag_a(Pw, 64);
    v16bf aP1 = load_frag_a(Pw + 32, 64);
#pragma unroll
    for (int jt = 0; jt < 4; ++jt) {
      O[jt] = wmma_bf16(aP0, load_frag_b(VsT + (jt * 16) * 64, 64), O[jt]);
      O[jt] = wmma_bf16(aP1, load_frag_b(VsT + (jt * 16) * 64 + 32, 64), O[jt]);
    }
  }

  // normalize and store merged-head output [B,T,D] bf16
  int bb = bh >> 4, hh = bh & 15;
#pragma unroll
  for (int r = 0; r < 8; ++r) {
    float inv = 1.0f / lrow[r];
    int qi = qw0 + r + 8 * hl;
#pragma unroll
    for (int jt = 0; jt < 4; ++jt)
      attn[((size_t)(bb * SEQ_T + qi)) * DMODEL + hh * DHEAD + jt * 16 + nlo] =
          (__bf16)(O[jt][r] * inv);
  }
}

// ---------------------------------------------------------------- launch
extern "C" void kernel_launch(void* const* d_in, const int* in_sizes, int n_in,
                              void* d_out, int out_size, void* d_ws,
                              size_t ws_size, hipStream_t stream) {
  const float* x      = (const float*)d_in[0];  // [2,2048,1024]
  const float* W_attn = (const float*)d_in[1];  // [1024,3072]
  const float* b_attn = (const float*)d_in[2];  // [3072]
  const float* W_proj = (const float*)d_in[3];  // [1024,1024]
  const float* b_proj = (const float*)d_in[4];  // [1024]
  float* out = (float*)d_out;

  char* ws = (char*)d_ws;
  __bf16* xb     = (__bf16*)(ws);                        // 8 MiB
  __bf16* wqkvb  = (__bf16*)(ws + ((size_t)8  << 20));   // 6 MiB
  __bf16* wprojb = (__bf16*)(ws + ((size_t)14 << 20));   // 2 MiB
  __bf16* qb     = (__bf16*)(ws + ((size_t)16 << 20));   // 8 MiB
  __bf16* kb     = (__bf16*)(ws + ((size_t)24 << 20));   // 8 MiB
  __bf16* vb     = (__bf16*)(ws + ((size_t)32 << 20));   // 8 MiB
  __bf16* attnb  = (__bf16*)(ws + ((size_t)40 << 20));   // 8 MiB

  int n;
  n = 2 * SEQ_T * DMODEL;
  f32_to_bf16_kernel<<<(n + 255) / 256, 256, 0, stream>>>(x, xb, n);
  n = DMODEL * 3 * DMODEL;
  f32_to_bf16_kernel<<<(n + 255) / 256, 256, 0, stream>>>(W_attn, wqkvb, n);
  n = DMODEL * DMODEL;
  f32_to_bf16_kernel<<<(n + 255) / 256, 256, 0, stream>>>(W_proj, wprojb, n);

  qkv_gemm_kernel<<<dim3(24, 32), 256, 0, stream>>>(xb, wqkvb, b_attn, qb, kb, vb);
  flash_attn_kernel<<<dim3(16, 32), 256, 0, stream>>>(qb, kb, vb, attnb);
  proj_gemm_kernel<<<dim3(8, 32), 256, 0, stream>>>(attnb, wprojb, b_proj, out);
}
